// BiRNN_11553462026468
// MI455X (gfx1250) — matrix operand order repaired and checked by reading, not compiled, not run on hardware
//
#include <hip/hip_runtime.h>
#include <hip/hip_bf16.h>

// Problem constants (match reference)
#define Bb 64
#define Tt 512
#define Ii 64
#define Hh 1024
#define Oo 512

typedef __bf16 bf16_t;
typedef __attribute__((ext_vector_type(16))) __bf16 v16bf;
typedef __attribute__((ext_vector_type(8)))  __bf16 v8bf;
typedef __attribute__((ext_vector_type(8)))  float  v8f;
typedef __attribute__((ext_vector_type(4)))  float  v4f;

__device__ __forceinline__ bf16_t f2bf(float f) { return (bf16_t)f; }
__device__ __forceinline__ float  bf2f(bf16_t b) { return (float)b; }

__device__ __forceinline__ v8f wmma_bf16(v16bf a, v16bf b, v8f c) {
    // D = A(16x32 bf16) * B(32x16 bf16) + C(16x16 f32)
    return __builtin_amdgcn_wmma_f32_16x16x32_bf16(
        false, a, false, b, (short)0, c, false, false);
}

// ---- Fragment loaders (layouts per CDNA5 ISA 7.12.2, wave32) ----

// A 16x32 bf16 from row-major bf16 (ld elements). lanes 0-15: M=lane, K base 0;
// lanes 16-31: M=lane-16, K base 8. elems 0-7 -> K=kb..kb+7, elems 8-15 -> K=kb+16..kb+23.
__device__ __forceinline__ v16bf load_a_bf16(const bf16_t* p0, int ld) {
    int lane = threadIdx.x & 31;
    int m  = lane & 15;
    int kb = (lane >> 4) << 3;
    const bf16_t* p = p0 + (size_t)m * ld + kb;
    v8bf lo = *(const v8bf*)(p);
    v8bf hi = *(const v8bf*)(p + 16);
    return __builtin_shufflevector(lo, hi, 0,1,2,3,4,5,6,7,8,9,10,11,12,13,14,15);
}

// A 16x32 bf16 from row-major fp32 (ld elements), converting on the fly.
__device__ __forceinline__ v16bf load_a_f32(const float* p0, int ld) {
    int lane = threadIdx.x & 31;
    int m  = lane & 15;
    int kb = (lane >> 4) << 3;
    const float* p = p0 + (size_t)m * ld + kb;
    v4f x0 = *(const v4f*)(p);
    v4f x1 = *(const v4f*)(p + 4);
    v4f x2 = *(const v4f*)(p + 16);
    v4f x3 = *(const v4f*)(p + 20);
    v16bf a;
#pragma unroll
    for (int e = 0; e < 4; ++e) {
        a[e]      = f2bf(x0[e]);
        a[e + 4]  = f2bf(x1[e]);
        a[e + 8]  = f2bf(x2[e]);
        a[e + 12] = f2bf(x3[e]);
    }
    return a;
}

// B 32x16 bf16 from K-transposed storage: element (k,n) at p0[n*ldT + k].
// lanes 0-15: N=lane, K=0..7,16..23; lanes 16-31: N=lane-16, K=8..15,24..31.
__device__ __forceinline__ v16bf load_b_bf16T(const bf16_t* p0, int ldT) {
    int lane = threadIdx.x & 31;
    int n  = lane & 15;
    int kb = (lane >> 4) << 3;
    const bf16_t* p = p0 + (size_t)n * ldT + kb;
    v8bf lo = *(const v8bf*)(p);
    v8bf hi = *(const v8bf*)(p + 16);
    return __builtin_shufflevector(lo, hi, 0,1,2,3,4,5,6,7,8,9,10,11,12,13,14,15);
}

// ---------------------------------------------------------------------------
// K0: init — zero barrier counters, convert h0 (fp32) -> bf16 state
// ---------------------------------------------------------------------------
__global__ void k0_init(const float* __restrict__ h0f, const float* __restrict__ h0b,
                        bf16_t* __restrict__ h0buf, unsigned* __restrict__ counters) {
    int i = blockIdx.x * blockDim.x + threadIdx.x;
    if (i < Bb * Hh) {
        h0buf[i]           = f2bf(h0f[i]);
        h0buf[Bb * Hh + i] = f2bf(h0b[i]);
    }
    if (i == 0) { counters[0] = 0u; counters[1] = 0u; }
}

// ---------------------------------------------------------------------------
// K1: xp[t,b,h] = sum_i x[b,t,i]*Wx[i,h] + bx[h]   (bf16 out)
// grid (H/64, T), 128 threads. Wx tile transposed in LDS.
// ---------------------------------------------------------------------------
__global__ void k1_xp(const float* __restrict__ x, const float* __restrict__ Wx,
                      const float* __restrict__ bx, bf16_t* __restrict__ xp) {
    __shared__ bf16_t wxT[64 * 64];   // [n][k]
    __shared__ float  bxs[64];
    int n0 = blockIdx.x * 64;
    int t  = blockIdx.y;

    for (int idx = threadIdx.x; idx < 64 * 64; idx += 128) {
        int n = idx >> 6, k = idx & 63;
        wxT[n * 64 + k] = f2bf(Wx[(size_t)k * Hh + n0 + n]);
    }
    if (threadIdx.x < 64) bxs[threadIdx.x] = bx[n0 + threadIdx.x];
    __syncthreads();

    int wave = threadIdx.x >> 5, lane = threadIdx.x & 31;
    int m0 = wave * 16;                       // rows = batch b
    v8f acc[4] = {v8f{}, v8f{}, v8f{}, v8f{}};

#pragma unroll
    for (int kk = 0; kk < Ii; kk += 32) {
        v16bf a = load_a_f32(x + (size_t)m0 * (Tt * Ii) + (size_t)t * Ii + kk, Tt * Ii);
#pragma unroll
        for (int nt = 0; nt < 4; ++nt) {
            v16bf b = load_b_bf16T(&wxT[(nt * 16) * 64 + kk], 64);
            acc[nt] = wmma_bf16(a, b, acc[nt]);
        }
    }

    int rbase = (lane >> 4) * 8, col = lane & 15;
#pragma unroll
    for (int nt = 0; nt < 4; ++nt) {
        int n = n0 + nt * 16 + col;
        float bias = bxs[nt * 16 + col];
#pragma unroll
        for (int r = 0; r < 8; ++r) {
            int b_ = m0 + rbase + r;
            xp[(size_t)t * Bb * Hh + (size_t)b_ * Hh + n] = f2bf(acc[nt][r] + bias);
        }
    }
}

// ---------------------------------------------------------------------------
// K2: bidirectional scan. 64 blocks = 2 dirs x 32 N-slices (32 cols each),
// 128 threads (4 waves). Ws slice lives transposed in LDS for the whole scan.
// Per step: h_t[:,slice] = sigmoid(xp_t + h_{t-1} @ Ws[:,slice] + bs), then
// release/acquire atomic grid barrier per direction.
// ---------------------------------------------------------------------------
__global__ void k2_scan(const float* __restrict__ Ws, const float* __restrict__ bs,
                        const bf16_t* __restrict__ xp, const bf16_t* __restrict__ h0buf,
                        bf16_t* __restrict__ hs_f, bf16_t* __restrict__ hs_b,
                        unsigned* __restrict__ counters) {
    __shared__ bf16_t wsT[32 * Hh];   // [n][k], 64 KB
    __shared__ float  bss[32];

    int dir = blockIdx.x >> 5;        // 0 = fwd, 1 = bwd
    int nb  = blockIdx.x & 31;
    int n0  = nb * 32;

    for (int idx = threadIdx.x; idx < 32 * Hh; idx += 128) {
        int n = idx >> 10, k = idx & (Hh - 1);
        wsT[n * Hh + k] = f2bf(Ws[(size_t)k * Hh + n0 + n]);
    }
    if (threadIdx.x < 32) bss[threadIdx.x] = bs[n0 + threadIdx.x];
    __syncthreads();

    bf16_t*   hs  = dir ? hs_b : hs_f;
    unsigned* cnt = counters + dir;

    int wave = threadIdx.x >> 5, lane = threadIdx.x & 31;
    int m0 = wave * 16;
    int rbase = (lane >> 4) * 8, col = lane & 15;

    for (int s = 0; s < Tt; ++s) {
        int t = dir ? (Tt - 1 - s) : s;

        // Prefetch next step's xp slice (read-once data coming from HBM):
        // hide its latency behind this step's GEMM.
        if (s + 1 < Tt) {
            int tn = dir ? (Tt - 2 - s) : (s + 1);
            int row = threadIdx.x;     // 128 threads; rows 0..63 useful
            if (row < Bb) {
                const bf16_t* pf = xp + (size_t)tn * Bb * Hh + (size_t)row * Hh + n0;
                __builtin_prefetch((const void*)pf, 0, 1);
            }
        }

        const bf16_t* hprev = (s == 0)
            ? (h0buf + (size_t)dir * Bb * Hh)
            : (hs + (size_t)(dir ? t + 1 : t - 1) * Bb * Hh);

        v8f acc0 = v8f{}, acc1 = v8f{};
#pragma unroll 4
        for (int kk = 0; kk < Hh; kk += 32) {
            v16bf a  = load_a_bf16(hprev + (size_t)m0 * Hh + kk, Hh);
            v16bf b0 = load_b_bf16T(&wsT[0 * Hh + kk], Hh);
            v16bf b1 = load_b_bf16T(&wsT[16 * Hh + kk], Hh);
            acc0 = wmma_bf16(a, b0, acc0);
            acc1 = wmma_bf16(a, b1, acc1);
        }

        bf16_t*       hout = hs + (size_t)t * Bb * Hh;
        const bf16_t* xpt  = xp + (size_t)t * Bb * Hh;
#pragma unroll
        for (int j = 0; j < 2; ++j) {
            const v8f& acc = j ? acc1 : acc0;
            int n = n0 + j * 16 + col;
            float bias = bss[j * 16 + col];
#pragma unroll
            for (int r = 0; r < 8; ++r) {
                int b_ = m0 + rbase + r;
                float v = acc[r] + bias + bf2f(xpt[(size_t)b_ * Hh + n]);
                v = 1.0f / (1.0f + __expf(-v));
                hout[(size_t)b_ * Hh + n] = f2bf(v);
            }
        }

        // grid barrier over this direction's 32 blocks
        __threadfence();
        __syncthreads();
        if (threadIdx.x == 0) {
            __hip_atomic_fetch_add(cnt, 1u, __ATOMIC_RELEASE, __HIP_MEMORY_SCOPE_AGENT);
            unsigned target = (unsigned)(s + 1) * 32u;
            while (__hip_atomic_load(cnt, __ATOMIC_ACQUIRE, __HIP_MEMORY_SCOPE_AGENT) < target)
                __builtin_amdgcn_s_sleep(1);
        }
        __syncthreads();
    }
}

// ---------------------------------------------------------------------------
// K3: out[b,t,:] = (hs_f[t,b,:] + hs_b[t,b,:]) @ Wout + bout
//               =  hs_f@Wout + hs_b@Wout   (two WMMAs per tile, no VALU add)
// grid (O/64, T*B/64), 128 threads. Wout slice transposed in LDS (128 KB).
// ---------------------------------------------------------------------------
__global__ void k3_out(const bf16_t* __restrict__ hs_f, const bf16_t* __restrict__ hs_b,
                       const float* __restrict__ Wout, const float* __restrict__ bout,
                       float* __restrict__ out) {
    __shared__ bf16_t woT[64 * Hh];   // [n][k], 128 KB
    __shared__ float  bo[64];
    int n0 = blockIdx.x * 64;
    int r0 = blockIdx.y * 64;         // flattened row r = t*64 + b

    for (int idx = threadIdx.x; idx < 64 * Hh; idx += 128) {
        int n = idx >> 10, k = idx & (Hh - 1);
        woT[n * Hh + k] = f2bf(Wout[(size_t)k * Oo + n0 + n]);
    }
    if (threadIdx.x < 64) bo[threadIdx.x] = bout[n0 + threadIdx.x];
    __syncthreads();

    int wave = threadIdx.x >> 5, lane = threadIdx.x & 31;
    int m0 = wave * 16;
    v8f acc[4] = {v8f{}, v8f{}, v8f{}, v8f{}};

    for (int kk = 0; kk < Hh; kk += 32) {
        v16bf af = load_a_bf16(hs_f + (size_t)(r0 + m0) * Hh + kk, Hh);
        v16bf ab = load_a_bf16(hs_b + (size_t)(r0 + m0) * Hh + kk, Hh);
#pragma unroll
        for (int nt = 0; nt < 4; ++nt) {
            v16bf b = load_b_bf16T(&woT[(nt * 16) * Hh + kk], Hh);
            acc[nt] = wmma_bf16(af, b, acc[nt]);
            acc[nt] = wmma_bf16(ab, b, acc[nt]);
        }
    }

    int rbase = (lane >> 4) * 8, col = lane & 15;
#pragma unroll
    for (int nt = 0; nt < 4; ++nt) {
        int o = n0 + nt * 16 + col;
        float bias = bo[nt * 16 + col];
#pragma unroll
        for (int rr = 0; rr < 8; ++rr) {
            int r  = r0 + m0 + rbase + rr;
            int b_ = r & 63;
            int t  = r >> 6;
            out[(size_t)b_ * Tt * Oo + (size_t)t * Oo + o] = acc[nt][rr] + bias;
        }
    }
}

// ---------------------------------------------------------------------------
extern "C" void kernel_launch(void* const* d_in, const int* in_sizes, int n_in,
                              void* d_out, int out_size, void* d_ws, size_t ws_size,
                              hipStream_t stream) {
    const float* x    = (const float*)d_in[0];
    const float* h0f  = (const float*)d_in[1];
    const float* h0b  = (const float*)d_in[2];
    const float* Wx   = (const float*)d_in[3];
    const float* bx   = (const float*)d_in[4];
    const float* Ws   = (const float*)d_in[5];
    const float* bs   = (const float*)d_in[6];
    const float* Wout = (const float*)d_in[7];
    const float* bout = (const float*)d_in[8];

    char* ws = (char*)d_ws;
    const size_t SZ_ACT = (size_t)Tt * Bb * Hh * sizeof(bf16_t);   // 64 MB
    bf16_t*   xp       = (bf16_t*)(ws);
    bf16_t*   hs_f     = (bf16_t*)(ws + SZ_ACT);
    bf16_t*   hs_b     = (bf16_t*)(ws + 2 * SZ_ACT);
    bf16_t*   h0buf    = (bf16_t*)(ws + 3 * SZ_ACT);
    unsigned* counters = (unsigned*)(ws + 3 * SZ_ACT + 2 * (size_t)Bb * Hh * sizeof(bf16_t) + 256);

    k0_init<<<(Bb * Hh + 255) / 256, 256, 0, stream>>>(h0f, h0b, h0buf, counters);
    k1_xp<<<dim3(Hh / 64, Tt), 128, 0, stream>>>(x, Wx, bx, xp);
    k2_scan<<<64, 128, 0, stream>>>(Ws, bs, xp, h0buf, hs_f, hs_b, counters);
    k3_out<<<dim3(Oo / 64, (Tt * Bb) / 64), 128, 0, stream>>>(hs_f, hs_b, Wout, bout, (float*)d_out);
}
